// SGuniNoduleNet_11312943858081
// MI455X (gfx1250) — matrix operand order repaired and checked by reading, not compiled, not run on hardware
//
#include <hip/hip_runtime.h>
#include <cstdint>
#include <cstddef>

typedef __attribute__((ext_vector_type(16))) _Float16 v16h;
typedef __attribute__((ext_vector_type(8)))  _Float16 v8h;
typedef __attribute__((ext_vector_type(8)))  float    v8f;

// ============================ device kernels ============================

__global__ void k_zero_h(_Float16* __restrict__ d, int n) {
  int i = threadIdx.x;
  if (i < n) d[i] = (_Float16)0.f;
}

// f32 [C][M] (channel-major) -> f16 [M][Cpad] (channels-last, zero padded)
__global__ void k_pack_act(const float* __restrict__ src, _Float16* __restrict__ dst,
                           int C, size_t M, int Cpad) {
  size_t tot = M * (size_t)Cpad;
  size_t i = (size_t)blockIdx.x * blockDim.x + threadIdx.x;
  if (i >= tot) return;
  int c = (int)(i % (size_t)Cpad);
  size_t m = i / (size_t)Cpad;
  dst[i] = (c < C) ? (_Float16)src[(size_t)c * M + m] : (_Float16)0.f;
}

// conv weights f32 [N][Cin][k3] -> f16 [Npad][k3*Cpad] with K-order (tap, ci)
__global__ void k_pack_wc(const float* __restrict__ src, _Float16* __restrict__ dst,
                          int N, int Cin, int k3, int Cpad, size_t tot) {
  size_t i = (size_t)blockIdx.x * blockDim.x + threadIdx.x;
  if (i >= tot) return;
  int Kp = k3 * Cpad;
  int kp = (int)(i % (size_t)Kp);
  int n  = (int)(i / (size_t)Kp);
  int tap = kp / Cpad;
  int c   = kp - tap * Cpad;
  float v = (n < N && c < Cin) ? src[((size_t)n * Cin + c) * k3 + tap] : 0.f;
  dst[i] = (_Float16)v;
}

// gemm weights f32 [N][K] -> f16 [Npad][Kpad]
__global__ void k_pack_w(const float* __restrict__ src, _Float16* __restrict__ dst,
                         int N, int K, int Kpad, size_t tot) {
  size_t i = (size_t)blockIdx.x * blockDim.x + threadIdx.x;
  if (i >= tot) return;
  int k = (int)(i % (size_t)Kpad);
  int n = (int)(i / (size_t)Kpad);
  float v = (n < N && k < K) ? src[(size_t)n * K + k] : 0.f;
  dst[i] = (_Float16)v;
}

__global__ void k_fold_bn(float* sc, float* bi, const float* g, const float* be,
                          const float* cb, int C) {
  int c = threadIdx.x;
  if (c < C) { sc[c] = g[c]; bi[c] = g[c] * cb[c] + be[c]; }
}
__global__ void k_fold_plain(float* sc, float* bi, const float* cb, int C) {
  int c = threadIdx.x;
  if (c < C) { sc[c] = 1.f; bi[c] = cb[c]; }
}

// Implicit-GEMM conv3d, WMMA f32_16x16x32_f16, K-order = (tap, ci), ci padded.
// X : f16 channels-last [Din][Hin][Win][Cpad];  Z: zero page (>= 128+24 halfs)
// Wp: f16 [Npad][k3*Cpad]
// Of: f32 [Cout][M] channel-major (optional); Oh: f16 [M][CpadOut] (optional)
// 4 waves/block; each wave computes a 16(M) x 32(N) tile (two accumulators).
template <int KS>
__global__ __launch_bounds__(128)
void k_wmma_conv3d(const _Float16* __restrict__ X, const _Float16* __restrict__ Wp,
                   const _Float16* __restrict__ Z,
                   const float* __restrict__ scale, const float* __restrict__ bias,
                   float* __restrict__ Of, _Float16* __restrict__ Oh,
                   int Cpad, int Cout, int Din, int Hin, int Win,
                   int Do, int Ho, int Wo, int stride, int pad,
                   int CpadOut, int relu) {
  const int lane = threadIdx.x & 31;
  const int wav  = threadIdx.x >> 5;
  const int hf   = lane >> 4;
  const int l16  = lane & 15;
  const int M    = Do * Ho * Wo;
  const int tm   = (blockIdx.x * 4 + wav) * 16;
  const int tn   = blockIdx.y * 32;
  if (tm >= M) return;

  v8f acc[2] = {};

  int m = tm + l16;
  bool mvalid = (m < M);
  int od = 0, oh = 0, ow = 0;
  if (mvalid) { ow = m % Wo; int t = m / Wo; oh = t % Ho; od = t / Ho; }

  const int K3C = KS * KS * KS * Cpad;
  const _Float16* w0 = Wp + (size_t)(tn + l16) * K3C;
  const _Float16* w1 = Wp + (size_t)(tn + 16 + l16) * K3C;

  int tap = 0;
  for (int kd = 0; kd < KS; kd++) {
    int id = od * stride + kd - pad;
    bool vd = mvalid && ((unsigned)id < (unsigned)Din);
    for (int kh = 0; kh < KS; kh++) {
      int ih = oh * stride + kh - pad;
      bool vh = vd && ((unsigned)ih < (unsigned)Hin);
      for (int kw = 0; kw < KS; kw++, tap++) {
        int iw = ow * stride + kw - pad;
        bool v = vh && ((unsigned)iw < (unsigned)Win);
        long voff = ((long)id * Hin + (long)ih) * (long)Win + (long)iw;
        const _Float16* src = v ? (X + (size_t)voff * Cpad) : Z;  // zero page if OOB
        const _Float16* s8 = src + hf * 16;
        const _Float16* wb0 = w0 + (size_t)tap * Cpad + hf * 16;
        const _Float16* wb1 = w1 + (size_t)tap * Cpad + hf * 16;
        for (int c0 = 0; c0 < Cpad; c0 += 32) {
          const v8h* p = (const v8h*)(s8 + c0);
          v8h lo = p[0];   // K offs [c0+hf*16,   +8)
          v8h hi = p[1];   // K offs [c0+hf*16+8, +8)
          v16h a = __builtin_shufflevector(lo, hi, 0, 1, 2, 3, 4, 5, 6, 7,
                                           8, 9, 10, 11, 12, 13, 14, 15);
          v16h b0 = *(const v16h*)(wb0 + c0);
          v16h b1 = *(const v16h*)(wb1 + c0);
          acc[0] = __builtin_amdgcn_wmma_f32_16x16x32_f16(false, a, false, b0,
                                                          (short)0, acc[0], false, false);
          acc[1] = __builtin_amdgcn_wmma_f32_16x16x32_f16(false, a, false, b1,
                                                          (short)0, acc[1], false, false);
        }
      }
    }
  }

#pragma unroll
  for (int hn = 0; hn < 2; hn++) {
    int co = tn + hn * 16 + l16;
    bool cov = (co < Cout);
    float sc = cov ? scale[co] : 0.f;
    float bi = cov ? bias[co] : 0.f;
#pragma unroll
    for (int r = 0; r < 8; r++) {
      int mm = tm + hf * 8 + r;
      if (mm < M) {
        float vv = cov ? (sc * acc[hn][r] + bi) : 0.f;
        if (relu) vv = fmaxf(vv, 0.f);
        if (Of && cov) Of[(size_t)co * M + mm] = vv;
        if (Oh && co < CpadOut) Oh[(size_t)mm * CpadOut + co] = (_Float16)vv;
      }
    }
  }
}

// WMMA GEMM: out[M][N] = act( A[M][Kpad] @ Wp[Npad][Kpad]^T + bias )
// (M is a multiple of 16 at all call sites; tail rows read row 0, discarded.)
__global__ __launch_bounds__(32)
void k_wmma_gemm(const _Float16* __restrict__ A, const _Float16* __restrict__ Wp,
                 const float* __restrict__ bias,
                 float* __restrict__ Of, _Float16* __restrict__ Oh,
                 int M, int N, int K, int Kpad, int relu) {
  const int lane = threadIdx.x;
  const int hf   = lane >> 4;
  const int l16  = lane & 15;
  const int tm   = blockIdx.x * 16;
  const int tn   = blockIdx.y * 16;
  v8f acc = {};
  int m = tm + l16;
  const _Float16* arow = A + (size_t)((m < M) ? m : 0) * Kpad + hf * 16;
  const _Float16* wrow = Wp + (size_t)(tn + l16) * Kpad + hf * 16;
  for (int kc = 0; kc < K; kc += 32) {
    const v8h* p = (const v8h*)(arow + kc);
    v8h lo = p[0];
    v8h hi = p[1];
    v16h afrag = __builtin_shufflevector(lo, hi, 0, 1, 2, 3, 4, 5, 6, 7,
                                         8, 9, 10, 11, 12, 13, 14, 15);
    v16h bfrag = *(const v16h*)(wrow + kc);
    acc = __builtin_amdgcn_wmma_f32_16x16x32_f16(false, afrag, false, bfrag,
                                                 (short)0, acc, false, false);
  }
  int n = tn + l16;
  if (n < N) {
    float bi = bias ? bias[n] : 0.f;
#pragma unroll
    for (int r = 0; r < 8; r++) {
      int mm = tm + hf * 8 + r;
      if (mm < M) {
        float v = acc[r] + bi;
        if (relu) v = fmaxf(v, 0.f);
        if (Of) Of[(size_t)mm * N + n] = v;
        if (Oh) Oh[(size_t)mm * N + n] = (_Float16)v;
      }
    }
  }
}

// 2x2x2 max pool stride 2; f32 channel-major out + f16 channels-last mirror
__global__ void k_maxpool2(const float* __restrict__ X, float* __restrict__ O,
                           _Float16* __restrict__ Oh, int C, int D, int Cpad) {
  int Do = D >> 1;
  size_t S = (size_t)Do * Do * Do;
  size_t tot = (size_t)C * S;
  size_t i = (size_t)blockIdx.x * blockDim.x + threadIdx.x;
  if (i >= tot) return;
  int w = (int)(i % Do); size_t t = i / Do;
  int h = (int)(t % Do); t /= Do;
  int d = (int)(t % Do); int c = (int)(t / Do);
  const float* b = X + (((size_t)c * D + 2 * d) * D + 2 * h) * D + 2 * w;
  float m = b[0];
  m = fmaxf(m, b[1]); m = fmaxf(m, b[D]); m = fmaxf(m, b[D + 1]);
  const float* b2 = b + (size_t)D * D;
  m = fmaxf(m, b2[0]); m = fmaxf(m, b2[1]); m = fmaxf(m, b2[D]); m = fmaxf(m, b2[D + 1]);
  O[i] = m;
  if (Oh) {
    size_t sp = i - (size_t)c * S;
    Oh[sp * Cpad + c] = (_Float16)m;
  }
}

// transposed conv 2x (einsum ncdhw,cokij->nodkhiwj) + bias + BN + relu (f32)
__global__ void k_deconv(const float* __restrict__ X, const float* __restrict__ Wt,
                         const float* __restrict__ B, const float* __restrict__ g,
                         const float* __restrict__ be, float* __restrict__ Out,
                         int C, int O, int d, int h, int w) {
  int D2 = 2 * d, H2 = 2 * h, W2 = 2 * w;
  size_t tot = (size_t)O * D2 * H2 * W2;
  size_t idx = (size_t)blockIdx.x * blockDim.x + threadIdx.x;
  if (idx >= tot) return;
  int ww = (int)(idx % W2); size_t t = idx / W2;
  int hh = (int)(t % H2); t /= H2;
  int dd = (int)(t % D2); int o = (int)(t / D2);
  int wbase = o * 8 + (dd & 1) * 4 + (hh & 1) * 2 + (ww & 1);
  int sd = dd >> 1, sh = hh >> 1, sw = ww >> 1;
  size_t xoff = ((size_t)sd * h + sh) * w + sw;
  size_t xstr = (size_t)d * h * w;
  float s = 0.f;
  for (int c = 0; c < C; c++)
    s += X[(size_t)c * xstr + xoff] * Wt[(size_t)c * O * 8 + wbase];
  float v = g[o] * (s + B[o]) + be[o];
  Out[idx] = fmaxf(v, 0.f);
}

// per-channel mean over M elements
__global__ void k_chan_mean(const float* __restrict__ X, float* __restrict__ mean,
                            int M) {
  int c = blockIdx.x;
  __shared__ float red[256];
  float s = 0.f;
  for (int i = threadIdx.x; i < M; i += 256) s += X[(size_t)c * M + i];
  red[threadIdx.x] = s;
  __syncthreads();
  for (int off = 128; off > 0; off >>= 1) {
    if (threadIdx.x < off) red[threadIdx.x] += red[threadIdx.x + off];
    __syncthreads();
  }
  if (threadIdx.x == 0) mean[c] = red[0] / (float)M;
}

// SE MLP: s = sigmoid(W2 @ relu(W1 @ mean + b1) + b2), single block
__global__ void k_se_mlp(const float* __restrict__ mean,
                         const float* __restrict__ w1, const float* __restrict__ b1,
                         const float* __restrict__ w2, const float* __restrict__ b2,
                         float* __restrict__ s, int C, int H) {
  __shared__ float mn[128];
  __shared__ float hid[16];
  int t = threadIdx.x;
  if (t < C) mn[t] = mean[t];
  __syncthreads();
  if (t < H) {
    float a = b1[t];
    for (int c = 0; c < C; c++) a += w1[t * C + c] * mn[c];
    hid[t] = fmaxf(a, 0.f);
  }
  __syncthreads();
  if (t < C) {
    float a = b2[t];
    for (int hh = 0; hh < H; hh++) a += w2[t * H + hh] * hid[hh];
    s[t] = 1.f / (1.f + expf(-a));
  }
}

// out = relu(x*s[c] + res); f32 channel-major + f16 channels-last mirror
__global__ void k_se_apply(const float* __restrict__ X, const float* __restrict__ s,
                           const float* __restrict__ res, float* __restrict__ O,
                           _Float16* __restrict__ Oh, int C, int M, int Cpad) {
  size_t tot = (size_t)C * M;
  size_t i = (size_t)blockIdx.x * blockDim.x + threadIdx.x;
  if (i >= tot) return;
  int c = (int)(i / (size_t)M);
  size_t m = i - (size_t)c * M;
  float v = X[i] * s[c] + res[i];
  v = fmaxf(v, 0.f);
  O[i] = v;
  if (Oh) Oh[m * Cpad + c] = (_Float16)v;
}

// transpose [C][M] -> [M][C]
__global__ void k_transpose_cm(const float* __restrict__ in, float* __restrict__ out,
                               int C, int M) {
  size_t tot = (size_t)C * M;
  size_t i = (size_t)blockIdx.x * blockDim.x + threadIdx.x;
  if (i >= tot) return;
  int c = (int)(i % (size_t)C);
  int m = (int)(i / (size_t)C);
  out[i] = in[(size_t)c * M + m];
}

// crop + adaptive 7x7x7 max pool over feat [64][24][24][24] -> A16 [48][Kpad]
__global__ void k_crop_pool(const float* __restrict__ feat, const int* __restrict__ props,
                            _Float16* __restrict__ A16, int Kpad) {
  const int CELLS = 343;
  size_t tot = (size_t)48 * 64 * CELLS;
  size_t idx = (size_t)blockIdx.x * blockDim.x + threadIdx.x;
  if (idx >= tot) return;
  int cell = (int)(idx % CELLS); size_t t = idx / CELLS;
  int c = (int)(t % 64); int p = (int)(t / 64);
  int kk = cell % 7, jj = (cell / 7) % 7, ii = cell / 49;
  const int* pr = props + p * 8;
  float cd = (float)pr[2], chh = (float)pr[3], cw = (float)pr[4];
  float sd = (float)pr[5], shh = (float)pr[6], sw = (float)pr[7];
  int d0 = (int)floorf((cd - sd * 0.5f) * 0.25f);   if (d0 < 0) d0 = 0;
  int h0 = (int)floorf((chh - shh * 0.5f) * 0.25f); if (h0 < 0) h0 = 0;
  int w0 = (int)floorf((cw - sw * 0.5f) * 0.25f);   if (w0 < 0) w0 = 0;
  int d1 = (int)ceilf((cd + sd * 0.5f) * 0.25f);    if (d1 > 24) d1 = 24;
  int h1 = (int)ceilf((chh + shh * 0.5f) * 0.25f);  if (h1 > 24) h1 = 24;
  int w1 = (int)ceilf((cw + sw * 0.5f) * 0.25f);    if (w1 > 24) w1 = 24;
  int nd = d1 - d0, nh = h1 - h0, nw = w1 - w0;
  int a0 = ii * nd / 7, a1 = ((ii + 1) * nd + 6) / 7;
  int b0 = jj * nh / 7, b1 = ((jj + 1) * nh + 6) / 7;
  int c0 = kk * nw / 7, c1 = ((kk + 1) * nw + 6) / 7;
  float m = -3.4e38f;
  for (int dz = a0; dz < a1; dz++)
    for (int dy = b0; dy < b1; dy++)
      for (int dx = c0; dx < c1; dx++)
        m = fmaxf(m, feat[(((size_t)c * 24 + (d0 + dz)) * 24 + (h0 + dy)) * 24 + (w0 + dx)]);
  A16[(size_t)p * Kpad + (size_t)c * CELLS + cell] = (_Float16)m;
}

// ============================ host side ============================

namespace {

inline int cpad32(int c) { return (c + 31) & ~31; }

struct Bump {
  char* base; size_t off;
  void* take(size_t b) { void* p = base + off; off = (off + b + 255) & ~(size_t)255; return p; }
};

struct BnP   { const float *be, *g; };
struct ConvP { const float *b, *w; };
struct BlockP {
  BnP bn1, bn2; ConvP c1, c2, sc; BnP scbn; bool has_sc;
  const float *seb1, *seb2, *sew1, *sew2;
};
struct PathP { const float* b; BnP bn; const float* w; };

struct Cursor {
  void* const* din; int i;
  const float* f() { return (const float*)din[i++]; }
};

BlockP readBlock(Cursor& c, bool sc) {
  BlockP p{};
  p.bn1.be = c.f(); p.bn1.g = c.f();
  p.bn2.be = c.f(); p.bn2.g = c.f();
  p.c1.b = c.f();   p.c1.w = c.f();
  p.c2.b = c.f();   p.c2.w = c.f();
  p.has_sc = sc;
  if (sc) { p.sc.b = c.f(); p.sc.w = c.f(); p.scbn.be = c.f(); p.scbn.g = c.f(); }
  p.seb1 = c.f(); p.seb2 = c.f(); p.sew1 = c.f(); p.sew2 = c.f();
  return p;
}

inline unsigned grids(size_t n) { return (unsigned)((n + 255) / 256); }

// xh: channels-last f16 with padded channel count Cpin
void conv_launch(hipStream_t st, const _Float16* xh, int Cin, int Cpin, int Din,
                 const float* w, const float* cb, const float* g, const float* be,
                 int Cout, int ks, int stride, int pad,
                 float* outf, _Float16* outh, int relu,
                 _Float16* wpack, const _Float16* zpage, float* scaleb, float* biasb) {
  int Do = (Din + 2 * pad - ks) / stride + 1;
  int k3 = ks * ks * ks;
  int CpOut = outh ? cpad32(Cout) : 0;
  int Npad = cpad32(Cout);               // multiple of 32 (two 16-wide n-tiles/wave)
  size_t wtot = (size_t)Npad * k3 * Cpin;
  k_pack_wc<<<grids(wtot), 256, 0, st>>>(w, wpack, Cout, Cin, k3, Cpin, wtot);
  if (g) k_fold_bn<<<1, 256, 0, st>>>(scaleb, biasb, g, be, cb, Cout);
  else   k_fold_plain<<<1, 256, 0, st>>>(scaleb, biasb, cb, Cout);
  int M = Do * Do * Do;
  dim3 grid((M + 63) / 64, Npad / 32);
  if (ks == 3)
    k_wmma_conv3d<3><<<grid, 128, 0, st>>>(xh, wpack, zpage, scaleb, biasb, outf, outh,
        Cpin, Cout, Din, Din, Din, Do, Do, Do, stride, pad, CpOut, relu);
  else
    k_wmma_conv3d<1><<<grid, 128, 0, st>>>(xh, wpack, zpage, scaleb, biasb, outf, outh,
        Cpin, Cout, Din, Din, Din, Do, Do, Do, stride, pad, CpOut, relu);
}

void gemm_launch(hipStream_t st, const _Float16* A, const float* w, const float* b,
                 int M, int N, int K, float* Of, _Float16* Oh, int relu,
                 _Float16* wpack) {
  int Kpad = (K + 31) & ~31;
  int Npad = (N + 15) & ~15;
  size_t wtot = (size_t)Npad * Kpad;
  k_pack_w<<<grids(wtot), 256, 0, st>>>(w, wpack, N, K, Kpad, wtot);
  dim3 grid((M + 15) / 16, (N + 15) / 16);
  k_wmma_gemm<<<grid, 32, 0, st>>>(A, wpack, b, Of, Oh, M, N, K, Kpad, relu);
}

void run_block(hipStream_t st, const BlockP& P, const float* xf, const _Float16* xh,
               int Cin, int Cout, int D, float* outf, _Float16* outh,
               _Float16* t1h, float* t2f, float* tresf,
               _Float16* wpack, const _Float16* zpage,
               float* scaleb, float* biasb, float* meanb, float* sb) {
  int M = D * D * D;
  int cpi = cpad32(Cin), cpo = cpad32(Cout);
  const float* res = xf;
  if (P.has_sc) {
    conv_launch(st, xh, Cin, cpi, D, P.sc.w, P.sc.b, P.scbn.g, P.scbn.be,
                Cout, 1, 1, 0, tresf, nullptr, 0, wpack, zpage, scaleb, biasb);
    res = tresf;
  }
  conv_launch(st, xh, Cin, cpi, D, P.c1.w, P.c1.b, P.bn1.g, P.bn1.be,
              Cout, 3, 1, 1, nullptr, t1h, 1, wpack, zpage, scaleb, biasb);
  conv_launch(st, t1h, Cout, cpo, D, P.c2.w, P.c2.b, P.bn2.g, P.bn2.be,
              Cout, 3, 1, 1, t2f, nullptr, 0, wpack, zpage, scaleb, biasb);
  k_chan_mean<<<Cout, 256, 0, st>>>(t2f, meanb, M);
  k_se_mlp<<<1, 128, 0, st>>>(meanb, P.sew1, P.seb1, P.sew2, P.seb2, sb, Cout, Cout / 16);
  size_t tot = (size_t)Cout * M;
  k_se_apply<<<grids(tot), 256, 0, st>>>(t2f, sb, res, outf, outh, Cout, M, cpo);
}

} // namespace

extern "C" void kernel_launch(void* const* d_in, const int* in_sizes, int n_in,
                              void* d_out, int out_size, void* d_ws, size_t ws_size,
                              hipStream_t stream) {
  (void)in_sizes; (void)n_in; (void)out_size;
  const float* x_in  = (const float*)d_in[0];
  const int*   props = (const int*)d_in[1];

  // ---- parse params in JAX pytree (alphabetical) order ----
  Cursor cu{d_in, 2};
  BlockP back2[3]; for (int i = 0; i < 3; i++) back2[i] = readBlock(cu, false);
  BlockP back3[3];
  back3[0] = readBlock(cu, true);
  back3[1] = readBlock(cu, false);
  back3[2] = readBlock(cu, false);
  ConvP deltaP; deltaP.b = cu.f(); deltaP.w = cu.f();
  ConvP fc1;    fc1.b = cu.f();    fc1.w = cu.f();
  ConvP fc2;    fc2.b = cu.f();    fc2.w = cu.f();
  BlockP forw1[2]; forw1[0] = readBlock(cu, true);  forw1[1] = readBlock(cu, false);
  BlockP forw2[2]; forw2[0] = readBlock(cu, true);  forw2[1] = readBlock(cu, false);
  BlockP forw3[3]; for (int i = 0; i < 3; i++) forw3[i] = readBlock(cu, false);
  BlockP forw4[3]; for (int i = 0; i < 3; i++) forw4[i] = readBlock(cu, false);
  ConvP logitP; logitP.b = cu.f(); logitP.w = cu.f();
  PathP path1;  path1.b = cu.f(); path1.bn.be = cu.f(); path1.bn.g = cu.f(); path1.w = cu.f();
  PathP path2;  path2.b = cu.f(); path2.bn.be = cu.f(); path2.bn.g = cu.f(); path2.w = cu.f();
  ConvP pre1; pre1.b = cu.f(); pre1.w = cu.f();
  ConvP pre2; pre2.b = cu.f(); pre2.w = cu.f();
  BnP prebn1; prebn1.be = cu.f(); prebn1.g = cu.f();
  BnP prebn2; prebn2.be = cu.f(); prebn2.g = cu.f();
  ConvP rpn_conv;   rpn_conv.b = cu.f();   rpn_conv.w = cu.f();
  ConvP rpn_deltas; rpn_deltas.b = cu.f(); rpn_deltas.w = cu.f();
  ConvP rpn_logits; rpn_logits.b = cu.f(); rpn_logits.w = cu.f();

  // ---- workspace layout ----
  const size_t S96 = (size_t)96 * 96 * 96;
  const size_t S48 = 48 * 48 * 48, S24 = 24 * 24 * 24, S12 = 12 * 12 * 12;
  const size_t GSZ  = 32 * S48;         // largest f32 chain tensor
  const size_t GSZH = 32 * S48;         // largest channels-last f16 tensor
  Bump ws{(char*)d_ws, 0};
  auto F = [&](size_t n) { return (float*)ws.take(n * sizeof(float)); };
  auto H = [&](size_t n) { return (_Float16*)ws.take(n * sizeof(_Float16)); };

  _Float16* x16 = H(S96 * 32);          // input, channels-last padded to 32
  float* cAf = F(GSZ); _Float16* cAh = H(GSZH);
  float* cBf = F(GSZ); _Float16* cBh = H(GSZH);
  _Float16* t1h = H(GSZH);
  float* t2f   = F(GSZ);
  float* tresf = F(GSZ);
  float* OUT2f = F(64 * S24);           // feat4, persists to crop stage
  float* OUT3f = F(64 * S12);
  _Float16* wpack = H((size_t)512 * 21952);   // largest packed weight (fc1)
  _Float16* zpage = H(256);                   // zero page for OOB conv taps
  float* scaleb = F(512); float* biasb = F(512);
  float* meanb = F(128);  float* sb = F(128);
  float* logT = F(3 * S24);
  float* delT = F(18 * S24);
  _Float16* crops16 = H((size_t)48 * 21952);
  _Float16* fc1h = H((size_t)48 * 512);
  _Float16* fc2h = H((size_t)48 * 256);
  if (ws.off > ws_size) return;

  float* dout = (float*)d_out;

  k_zero_h<<<1, 256, 0, stream>>>(zpage, 256);

  // ---- backbone ----
  k_pack_act<<<grids(S96 * 32), 256, 0, stream>>>(x_in, x16, 1, S96, 32);

  conv_launch(stream, x16, 1, 32, 96, pre1.w, pre1.b, prebn1.g, prebn1.be,
              24, 3, 2, 1, cAf, cAh, 1, wpack, zpage, scaleb, biasb);          // 24 @ 48^3
  conv_launch(stream, cAh, 24, 32, 48, pre2.w, pre2.b, prebn2.g, prebn2.be,
              24, 3, 1, 1, cBf, cBh, 1, wpack, zpage, scaleb, biasb);

  run_block(stream, forw1[0], cBf, cBh, 24, 32, 48, cAf, cAh, t1h, t2f, tresf, wpack, zpage, scaleb, biasb, meanb, sb);
  run_block(stream, forw1[1], cAf, cAh, 32, 32, 48, cBf, cBh, t1h, t2f, tresf, wpack, zpage, scaleb, biasb, meanb, sb);

  k_maxpool2<<<grids(32 * S24), 256, 0, stream>>>(cBf, cAf, cAh, 32, 48, 32);
  run_block(stream, forw2[0], cAf, cAh, 32, 64, 24, cBf, cBh, t1h, t2f, tresf, wpack, zpage, scaleb, biasb, meanb, sb);
  run_block(stream, forw2[1], cBf, cBh, 64, 64, 24, OUT2f, nullptr, t1h, t2f, tresf, wpack, zpage, scaleb, biasb, meanb, sb);

  k_maxpool2<<<grids(64 * S12), 256, 0, stream>>>(OUT2f, cAf, cAh, 64, 24, 64);
  run_block(stream, forw3[0], cAf, cAh, 64, 64, 12, cBf, cBh, t1h, t2f, tresf, wpack, zpage, scaleb, biasb, meanb, sb);
  run_block(stream, forw3[1], cBf, cBh, 64, 64, 12, cAf, cAh, t1h, t2f, tresf, wpack, zpage, scaleb, biasb, meanb, sb);
  run_block(stream, forw3[2], cAf, cAh, 64, 64, 12, OUT3f, nullptr, t1h, t2f, tresf, wpack, zpage, scaleb, biasb, meanb, sb);

  k_maxpool2<<<grids(64 * 216), 256, 0, stream>>>(OUT3f, cAf, cAh, 64, 12, 64);
  run_block(stream, forw4[0], cAf, cAh, 64, 64, 6, cBf, cBh, t1h, t2f, tresf, wpack, zpage, scaleb, biasb, meanb, sb);
  run_block(stream, forw4[1], cBf, cBh, 64, 64, 6, cAf, cAh, t1h, t2f, tresf, wpack, zpage, scaleb, biasb, meanb, sb);
  run_block(stream, forw4[2], cAf, cAh, 64, 64, 6, cBf, cBh, t1h, t2f, tresf, wpack, zpage, scaleb, biasb, meanb, sb);
  // out4 = cBf (64 @ 6^3)

  // ---- decoder: comb3 = [deconv(out4), out3] ----
  k_deconv<<<grids(64 * S12), 256, 0, stream>>>(cBf, path1.w, path1.b,
      path1.bn.g, path1.bn.be, cAf, 64, 64, 6, 6, 6);
  hipMemcpyAsync(cAf + (size_t)64 * S12, OUT3f, (size_t)64 * S12 * sizeof(float),
                 hipMemcpyDeviceToDevice, stream);
  k_pack_act<<<grids(S12 * 128), 256, 0, stream>>>(cAf, cAh, 128, S12, 128);

  run_block(stream, back3[0], cAf, cAh, 128, 64, 12, cBf, cBh, t1h, t2f, tresf, wpack, zpage, scaleb, biasb, meanb, sb);
  run_block(stream, back3[1], cBf, cBh, 64, 64, 12, cAf, cAh, t1h, t2f, tresf, wpack, zpage, scaleb, biasb, meanb, sb);
  run_block(stream, back3[2], cAf, cAh, 64, 64, 12, cBf, cBh, t1h, t2f, tresf, wpack, zpage, scaleb, biasb, meanb, sb);

  // comb2 = [deconv(comb3), out2]
  k_deconv<<<grids(64 * S24), 256, 0, stream>>>(cBf, path2.w, path2.b,
      path2.bn.g, path2.bn.be, cAf, 64, 64, 12, 12, 12);
  hipMemcpyAsync(cAf + (size_t)64 * S24, OUT2f, (size_t)64 * S24 * sizeof(float),
                 hipMemcpyDeviceToDevice, stream);
  k_pack_act<<<grids(S24 * 128), 256, 0, stream>>>(cAf, cAh, 128, S24, 128);

  run_block(stream, back2[0], cAf, cAh, 128, 128, 24, cBf, cBh, t1h, t2f, tresf, wpack, zpage, scaleb, biasb, meanb, sb);
  run_block(stream, back2[1], cBf, cBh, 128, 128, 24, cAf, cAh, t1h, t2f, tresf, wpack, zpage, scaleb, biasb, meanb, sb);
  run_block(stream, back2[2], cAf, cAh, 128, 128, 24, cBf, cBh, t1h, t2f, tresf, wpack, zpage, scaleb, biasb, meanb, sb);
  // fs = cB (128 @ 24^3)

  // ---- RPN heads (1x1x1 convs via WMMA) ----
  conv_launch(stream, cBh, 128, 128, 24, rpn_conv.w, rpn_conv.b, nullptr, nullptr,
              64, 1, 1, 0, nullptr, t1h, 1, wpack, zpage, scaleb, biasb);
  conv_launch(stream, t1h, 64, 64, 24, rpn_logits.w, rpn_logits.b, nullptr, nullptr,
              3, 1, 1, 0, logT, nullptr, 0, wpack, zpage, scaleb, biasb);
  conv_launch(stream, t1h, 64, 64, 24, rpn_deltas.w, rpn_deltas.b, nullptr, nullptr,
              18, 1, 1, 0, delT, nullptr, 0, wpack, zpage, scaleb, biasb);
  k_transpose_cm<<<grids(3 * S24), 256, 0, stream>>>(logT, dout, 3, (int)S24);
  k_transpose_cm<<<grids(18 * S24), 256, 0, stream>>>(delT, dout + 41472, 18, (int)S24);

  // ---- RCNN head: crop + adaptive max pool + FCs ----
  k_crop_pool<<<grids((size_t)48 * 64 * 343), 256, 0, stream>>>(OUT2f, props, crops16, 21952);
  gemm_launch(stream, crops16, fc1.w, fc1.b, 48, 512, 21952, nullptr, fc1h, 1, wpack);
  gemm_launch(stream, fc1h,    fc2.w, fc2.b, 48, 256, 512,   nullptr, fc2h, 1, wpack);
  gemm_launch(stream, fc2h, logitP.w, logitP.b, 48, 2,  256, dout + 290304, nullptr, 0, wpack);
  gemm_launch(stream, fc2h, deltaP.w, deltaP.b, 48, 12, 256, dout + 290400, nullptr, 0, wpack);
}
